// LSTM_Decoder_21234318311669
// MI455X (gfx1250) — matrix operand order repaired and checked by reading, not compiled, not run on hardware
//
#include <hip/hip_runtime.h>
#include <hip/hip_bf16.h>
#include <math.h>

// ---- problem constants (match reference) ----
#define BATCH 64
#define SEQ   256
#define INPUT 1024
#define HID   1024
#define GATES 4096   // 4*H, order i,f,g,o

typedef __attribute__((ext_vector_type(16))) _Float16 v16h;
typedef __attribute__((ext_vector_type(8)))  _Float16 v8h;
typedef __attribute__((ext_vector_type(8)))  float    v8f;

union AFrag { v16h v; v8h h[2]; };

__device__ __forceinline__ float sigmoidf_(float x) {
  return 1.0f / (1.0f + expf(-x));
}

// ============================================================================
// Fused LSTM layer step (one wave = one 16x16 (batch, hidden) block):
//   gates[64,4096] = A1[64,K1] @ W1[4096,K1]^T + Hprev[64,1024] @ W2^T + bias
//   then in-register: i,f,g,o -> c_new, h_new; writes c, h(f32), h(f16), out.
// The wave holds the four gate tiles at N = {n0, H+n0, 2H+n0, 3H+n0}, sharing
// a single A fragment per K-chunk (4 WMMAs per A load).
// WMMA operands use contiguous k order (k = 16*half + s) for A and B: per-lane A = two contiguous 8-half
// chunks of row m0+(lane&15); per-lane B = one contiguous 16-half chunk of
// W row n+(lane&15) (B[k][n] = W[n][k]; K contiguous in W rows).
// h16in/h16out are ping-pong buffers: the GEMM reads ALL columns of h16in
// while h16out is written only at owned elements -> no intra-launch race.
// ============================================================================
__global__ __launch_bounds__(256) void lstm_layer_kernel(
    const _Float16* __restrict__ A1, long lda1,
    const _Float16* __restrict__ W1, int K1,
    const _Float16* __restrict__ h16in,
    const _Float16* __restrict__ W2,
    const float*    __restrict__ bias,   // [4096] = b_ih + b_hh
    float*          __restrict__ c,      // [64,1024] state (exclusive elems)
    float*          __restrict__ h,      // [64,1024] f32 state (exclusive)
    _Float16*       __restrict__ h16out, // [64,1024] ping-pong partner
    float*          __restrict__ out)    // nullable; pre-offset by s*H
{
  const int gwave = (blockIdx.x * 256 + threadIdx.x) >> 5;  // 0..255
  const int lane  = threadIdx.x & 31;
  const int mtile = gwave & 3;          // 64 rows   -> 4 M-tiles
  const int ncol  = gwave >> 2;         // 1024 cols -> 64 N-positions
  const int m0   = mtile << 4;
  const int n0   = ncol << 4;
  const int r    = lane & 15;
  const int half = lane >> 4;

  v8f acc_i = {}, acc_f = {}, acc_g = {}, acc_o = {};

  // ---- A1 @ W1^T (input path) ----
  {
    const _Float16* aRow = A1 + (long)(m0 + r) * lda1;
    const _Float16* bI = W1 + (long)(0 * HID + n0 + r) * (long)K1;
    const _Float16* bF = W1 + (long)(1 * HID + n0 + r) * (long)K1;
    const _Float16* bG = W1 + (long)(2 * HID + n0 + r) * (long)K1;
    const _Float16* bO = W1 + (long)(3 * HID + n0 + r) * (long)K1;
#pragma unroll 2
    for (int k = 0; k < K1; k += 32) {
      AFrag a;
      a.h[0] = *(const v8h*)(aRow + k + half * 16);
      a.h[1] = *(const v8h*)(aRow + k + half * 16 + 8);
      v16h vi = *(const v16h*)(bI + k + half * 16);
      v16h vf = *(const v16h*)(bF + k + half * 16);
      v16h vg = *(const v16h*)(bG + k + half * 16);
      v16h vo = *(const v16h*)(bO + k + half * 16);
      acc_i = __builtin_amdgcn_wmma_f32_16x16x32_f16(false, a.v, false, vi, (short)0, acc_i, false, false);
      acc_f = __builtin_amdgcn_wmma_f32_16x16x32_f16(false, a.v, false, vf, (short)0, acc_f, false, false);
      acc_g = __builtin_amdgcn_wmma_f32_16x16x32_f16(false, a.v, false, vg, (short)0, acc_g, false, false);
      acc_o = __builtin_amdgcn_wmma_f32_16x16x32_f16(false, a.v, false, vo, (short)0, acc_o, false, false);
    }
  }
  // ---- Hprev @ W2^T (recurrent path, K = HID) ----
  {
    const _Float16* aRow = h16in + (long)(m0 + r) * HID;
    const _Float16* bI = W2 + (long)(0 * HID + n0 + r) * HID;
    const _Float16* bF = W2 + (long)(1 * HID + n0 + r) * HID;
    const _Float16* bG = W2 + (long)(2 * HID + n0 + r) * HID;
    const _Float16* bO = W2 + (long)(3 * HID + n0 + r) * HID;
#pragma unroll 2
    for (int k = 0; k < HID; k += 32) {
      AFrag a;
      a.h[0] = *(const v8h*)(aRow + k + half * 16);
      a.h[1] = *(const v8h*)(aRow + k + half * 16 + 8);
      v16h vi = *(const v16h*)(bI + k + half * 16);
      v16h vf = *(const v16h*)(bF + k + half * 16);
      v16h vg = *(const v16h*)(bG + k + half * 16);
      v16h vo = *(const v16h*)(bO + k + half * 16);
      acc_i = __builtin_amdgcn_wmma_f32_16x16x32_f16(false, a.v, false, vi, (short)0, acc_i, false, false);
      acc_f = __builtin_amdgcn_wmma_f32_16x16x32_f16(false, a.v, false, vf, (short)0, acc_f, false, false);
      acc_g = __builtin_amdgcn_wmma_f32_16x16x32_f16(false, a.v, false, vg, (short)0, acc_g, false, false);
      acc_o = __builtin_amdgcn_wmma_f32_16x16x32_f16(false, a.v, false, vo, (short)0, acc_o, false, false);
    }
  }

  // ---- fused LSTM cell epilogue ----
  // C layout: VGPR j -> (M = m0 + half*8 + j, N = n0 + r)
  const int hh = n0 + r;
  const float bi = bias[0 * HID + hh];
  const float bf = bias[1 * HID + hh];
  const float bg = bias[2 * HID + hh];
  const float bo = bias[3 * HID + hh];

#pragma unroll
  for (int j = 0; j < 8; ++j) {
    const int  m   = m0 + half * 8 + j;
    const long idx = (long)m * HID + hh;
    const float iv = sigmoidf_(acc_i[j] + bi);
    const float fv = sigmoidf_(acc_f[j] + bf);
    const float gv = tanhf(acc_g[j] + bg);
    const float ov = sigmoidf_(acc_o[j] + bo);
    const float cn = fv * c[idx] + iv * gv;
    const float hn = ov * tanhf(cn);
    c[idx]      = cn;
    h[idx]      = hn;
    h16out[idx] = (_Float16)hn;
    if (out) out[(long)m * (SEQ * HID) + hh] = hn;
  }
}

// ============================================================================
// Small setup kernels
// ============================================================================
__global__ __launch_bounds__(256) void f32_to_f16_kernel(
    const float* __restrict__ src, _Float16* __restrict__ dst, int n)
{
  const int idx = blockIdx.x * 256 + threadIdx.x;
  if (idx < n) dst[idx] = (_Float16)src[idx];
}

__global__ __launch_bounds__(256) void bias_sum_kernel(
    const float* __restrict__ a, const float* __restrict__ b,
    float* __restrict__ out, int n)
{
  const int idx = blockIdx.x * 256 + threadIdx.x;
  if (idx < n) out[idx] = a[idx] + b[idx];
}

__global__ __launch_bounds__(256) void init_state_kernel(
    const float* __restrict__ eh, const float* __restrict__ ec,
    float* __restrict__ h, float* __restrict__ c,
    _Float16* __restrict__ h16, int n)
{
  const int idx = blockIdx.x * 256 + threadIdx.x;
  if (idx < n) {
    const float hv = eh[idx];
    h[idx] = hv;
    c[idx] = ec[idx];
    h16[idx] = (_Float16)hv;
  }
}

__global__ __launch_bounds__(256) void copy_states_kernel(
    const float* __restrict__ h0, const float* __restrict__ c0,
    const float* __restrict__ h1, const float* __restrict__ c1,
    float* __restrict__ outh, float* __restrict__ outc)
{
  const int idx = blockIdx.x * 256 + threadIdx.x;   // < 64*1024
  outh[idx]                = h0[idx];
  outh[BATCH * HID + idx]  = h1[idx];
  outc[idx]                = c0[idx];
  outc[BATCH * HID + idx]  = c1[idx];
}

// ============================================================================
extern "C" void kernel_launch(void* const* d_in, const int* in_sizes, int n_in,
                              void* d_out, int out_size, void* d_ws, size_t ws_size,
                              hipStream_t stream) {
  const float* x    = (const float*)d_in[0];   // [B, S, IN]
  const float* eh   = (const float*)d_in[1];   // [L, B, H]
  const float* ec   = (const float*)d_in[2];   // [L, B, H]
  const float* Wih0 = (const float*)d_in[3];   // [4H, IN]
  const float* Whh0 = (const float*)d_in[4];   // [4H, H]
  const float* bih0 = (const float*)d_in[5];
  const float* bhh0 = (const float*)d_in[6];
  const float* Wih1 = (const float*)d_in[7];   // [4H, H]
  const float* Whh1 = (const float*)d_in[8];   // [4H, H]
  const float* bih1 = (const float*)d_in[9];
  const float* bhh1 = (const float*)d_in[10];
  float* out = (float*)d_out;

  // ---- workspace carve-up (≈66 MB) ----
  char* ws = (char*)d_ws;
  size_t off = 0;
  auto take = [&](size_t bytes) -> char* {
    char* p = ws + off;
    off = (off + bytes + 255) & ~(size_t)255;
    return p;
  };
  _Float16* Wih0h = (_Float16*)take((size_t)GATES * INPUT * 2);
  _Float16* Whh0h = (_Float16*)take((size_t)GATES * HID * 2);
  _Float16* Wih1h = (_Float16*)take((size_t)GATES * HID * 2);
  _Float16* Whh1h = (_Float16*)take((size_t)GATES * HID * 2);
  _Float16* x16   = (_Float16*)take((size_t)BATCH * SEQ * INPUT * 2);
  float*    bias0 = (float*)take(GATES * 4);
  float*    bias1 = (float*)take(GATES * 4);
  float*    h0    = (float*)take(BATCH * HID * 4);
  float*    c0    = (float*)take(BATCH * HID * 4);
  float*    h1    = (float*)take(BATCH * HID * 4);
  float*    c1    = (float*)take(BATCH * HID * 4);
  _Float16* h0_16a = (_Float16*)take(BATCH * HID * 2);   // ping-pong A
  _Float16* h0_16b = (_Float16*)take(BATCH * HID * 2);   // ping-pong B
  _Float16* h1_16a = (_Float16*)take(BATCH * HID * 2);
  _Float16* h1_16b = (_Float16*)take(BATCH * HID * 2);

  const int nW = GATES * INPUT;        // 4,194,304 per weight matrix
  const int nX = BATCH * SEQ * INPUT;  // 16,777,216
  const int nS = BATCH * HID;          // 65,536

  // ---- one-time conversions (f32 -> f16, bias fold, state init) ----
  f32_to_f16_kernel<<<nW / 256, 256, 0, stream>>>(Wih0, Wih0h, nW);
  f32_to_f16_kernel<<<nW / 256, 256, 0, stream>>>(Whh0, Whh0h, nW);
  f32_to_f16_kernel<<<nW / 256, 256, 0, stream>>>(Wih1, Wih1h, nW);
  f32_to_f16_kernel<<<nW / 256, 256, 0, stream>>>(Whh1, Whh1h, nW);
  f32_to_f16_kernel<<<nX / 256, 256, 0, stream>>>(x, x16, nX);
  bias_sum_kernel<<<GATES / 256, 256, 0, stream>>>(bih0, bhh0, bias0, GATES);
  bias_sum_kernel<<<GATES / 256, 256, 0, stream>>>(bih1, bhh1, bias1, GATES);
  init_state_kernel<<<nS / 256, 256, 0, stream>>>(eh,      ec,      h0, c0, h0_16a, nS);
  init_state_kernel<<<nS / 256, 256, 0, stream>>>(eh + nS, ec + nS, h1, c1, h1_16a, nS);

  // ---- recurrence over time (2 fused launches per step) ----
  // 256 waves per launch -> 32 blocks of 256 threads (8 waves).
  for (int s = 0; s < SEQ; ++s) {
    const int p = s & 1;
    _Float16* h0in  = p ? h0_16b : h0_16a;
    _Float16* h0out = p ? h0_16a : h0_16b;
    _Float16* h1in  = p ? h1_16b : h1_16a;
    _Float16* h1out = p ? h1_16a : h1_16b;

    // layer 0: input = x_t (strided rows), recurrent = h0in
    lstm_layer_kernel<<<32, 256, 0, stream>>>(
        x16 + (long)s * INPUT, (long)SEQ * INPUT, Wih0h, INPUT,
        h0in, Whh0h, bias0, c0, h0, h0out, nullptr);

    // layer 1: input = h0out (just written), recurrent = h1in
    lstm_layer_kernel<<<32, 256, 0, stream>>>(
        h0out, (long)HID, Wih1h, HID,
        h1in, Whh1h, bias1, c1, h1, h1out, out + (long)s * HID);
  }

  // ---- final (h, c) stacked after out[B,S,H] ----
  float* outh = out + (size_t)BATCH * SEQ * HID;
  float* outc = outh + (size_t)2 * BATCH * HID;   // L = 2
  copy_states_kernel<<<nS / 256, 256, 0, stream>>>(h0, c0, h1, c1, outh, outc);
}
